// CIN_88201448391488
// MI455X (gfx1250) — matrix operand order repaired and checked by reading, not compiled, not run on hardware
//
#include <hip/hip_runtime.h>

// ---------------------------------------------------------------------------
// Fused xDeepFM CIN (3 layers) for gfx1250 (MI455X, wave32, WMMA).
//   x  : [B=4096, m=39, D=64]  f32
//   W0 : [1521,128] f32   W1/W2 : [4992,128] f32
//   out: [B, 384] f32  (d-sums of h1 | h2 | h3)
//
// Formulation:  z[h,d] = sum_i x0[i,d] * Y_i[h,d],
//               Y_i[h,d] = sum_j W[i,j,h] * xk[j,d]   (pure GEMM, B = xk).
// xk fragments are i-invariant -> hoisted out of the i-loop (loaded once).
// Pass 1 (tiny): transpose + f16-convert weights into d_ws (Wt[h][k]; layer-0
//   per-field K padded 39->64 with zeros baked in).  ws need: 3,194,880 B.
// Pass 2: one workgroup per batch element; h1/h2 in LDS (f16);
//   v_wmma_f32_16x16x32_f16 core; f32 x0-scaling via v_fmac_f32;
//   d-sums via ds_swizzle_b32 xor-tree + ds_add_f32.
// ---------------------------------------------------------------------------

typedef __attribute__((ext_vector_type(16))) _Float16 v16h;
typedef __attribute__((ext_vector_type(8)))  _Float16 v8h;
typedef __attribute__((ext_vector_type(2)))  _Float16 v2h;
typedef __attribute__((ext_vector_type(8)))  float    v8f;

#define M_FIELDS 39
#define EMB      64
#define HN       128
#define XSTRIDE  52      // LDS stride of xT [64][52] f32 (x0 factors)
#define XSTRH    72      // LDS stride of xF [64][72] f16; cols 39..71 zeroed
#define HSTRIDE  136     // LDS stride of hT [64][136] f16 (mult of 8 -> 16B align)
#define THREADS  512     // 16 waves

// layer-0 padded geometry: k = i*64 + j, j in [0,64), pads are zero
#define HKP0   64
#define KP0    (M_FIELDS * HKP0)   // 2496
#define K12    (M_FIELDS * HN)     // 4992

// Wt regions inside d_ws (element offsets, all multiples of 8 halves)
#define WT_OFF0  0
#define WT_OFF1  (KP0 * HN)                  // 319488
#define WT_OFF2  (WT_OFF1 + K12 * HN)        // 958464
#define WT_BYTES ((WT_OFF2 + K12 * HN) * 2)  // 3,194,880

__device__ __forceinline__ v8f wmma_f16(v16h a, v16h b, v8f c) {
    return __builtin_amdgcn_wmma_f32_16x16x32_f16(
        /*neg_a=*/false, a, /*neg_b=*/false, b,
        /*c_mod=*/(short)0, c, /*reuse_a=*/false, /*reuse_b=*/false);
}

__device__ __forceinline__ v16h cat16(v8h a, v8h b) {
    return __builtin_shufflevector(a, b, 0, 1, 2, 3, 4, 5, 6, 7,
                                         8, 9, 10, 11, 12, 13, 14, 15);
}

// xor-butterfly add via DS_SWIZZLE_B32 (group-of-32: and=0x1f, xor in [14:10])
template<int IMM>
__device__ __forceinline__ float swz_add(float s) {
    const int t = __builtin_amdgcn_ds_swizzle(__float_as_int(s), IMM);
    return s + __int_as_float(t);
}

// ---------------------------------------------------------------------------
// Weight transpose/convert kernels (run once per launch; negligible cost).
// ---------------------------------------------------------------------------
__global__ void wt0_kernel(const float* __restrict__ W0, _Float16* __restrict__ Wt0) {
    const int idx = blockIdx.x * 256 + threadIdx.x;   // over KP0*HN
    if (idx >= KP0 * HN) return;
    const int h  = idx % HN;          // consecutive h -> coalesced W0 reads
    const int kp = idx / HN;
    const int i  = kp / HKP0;
    const int j  = kp % HKP0;
    float v = 0.f;
    if (j < M_FIELDS) v = W0[(i * M_FIELDS + j) * HN + h];
    Wt0[h * KP0 + kp] = (_Float16)v;
}

__global__ void wt12_kernel(const float* __restrict__ W, _Float16* __restrict__ Wt) {
    const int idx = blockIdx.x * 256 + threadIdx.x;   // over K12*HN
    if (idx >= K12 * HN) return;
    const int h = idx % HN;
    const int k = idx / HN;
    Wt[h * K12 + k] = (_Float16)W[idx];               // W[k*HN + h], coalesced read
}

// Store 16x16 f32 tile to LDS (f16, transposed) and reduce over d into osum.
// C layout: VGPR r, lanes 0-15 -> (M=r, N=lane), lanes 16-31 -> (M=r+8, N-16).
template<bool STORE_H>
__device__ __forceinline__ void finish_tile(v8f c, _Float16* hOut, float* osum,
                                            int h0, int d0, int lane) {
    const int laneN  = lane & 15;
    const int hiHalf = lane >> 4;
    const int d      = d0 + laneN;
    const int rbase  = h0 + hiHalf * 8;

    if (STORE_H) {
#pragma unroll
        for (int r = 0; r < 8; r += 2) {       // rows r,r+1 adjacent in hT[d][.]
            v2h p;
            p[0] = (_Float16)c[r];
            p[1] = (_Float16)c[r + 1];
            *(v2h*)(hOut + d * HSTRIDE + rbase + r) = p;   // ds_store_b32
        }
    }

    float sums[8];
#pragma unroll
    for (int r = 0; r < 8; ++r) {              // xor-tree over the 16 d-columns
        float s = c[r];
        s = swz_add<0x041F>(s);                // SWAPX1
        s = swz_add<0x081F>(s);                // SWAPX2
        s = swz_add<0x101F>(s);                // SWAPX4
        s = swz_add<0x201F>(s);                // SWAPX8
        sums[r] = s;
    }
    if (laneN == 0) {                          // one exec-masked region
#pragma unroll
        for (int r = 0; r < 8; ++r)
            atomicAdd(&osum[rbase + r], sums[r]);   // ds_add_f32
    }
}

// ---------------------------------------------------------------------------
// One CIN layer.  HKB = 32-K blocks per field (2 or 4); KS = Wt row stride.
// ---------------------------------------------------------------------------
template<int HKB, int KS, int XKSTRIDE, bool STORE_H>
__device__ __forceinline__ void cin_layer(const _Float16* __restrict__ Wt,
                                          const float* xT, const _Float16* xkT,
                                          _Float16* hOut, float* osum, int tid) {
    const int lane   = tid & 31;
    const int wave   = tid >> 5;          // 0..15
    const int ht     = wave & 7;          // h tile 0..7
    const int dt0    = wave >> 3;         // 0..1 (second tile = dt0+2)
    const int h0     = ht * 16;
    const int laneN  = lane & 15;
    const int hiHalf = lane >> 4;
    const int d0a    = dt0 * 16;
    const int d0b    = (dt0 + 2) * 16;

    // A: element t<8 -> K = kb + 8*hiHalf + t; t>=8 -> +16.  (ISA A layout)
    const _Float16* wrow = Wt + (h0 + laneN) * KS + hiHalf * 8;
    // B: lane column d; element t<8 -> K = kb + 8*hiHalf + t; t>=8 -> +16.
    const _Float16* xka  = xkT + (d0a + laneN) * XKSTRIDE + hiHalf * 8;
    const _Float16* xkb  = xkT + (d0b + laneN) * XKSTRIDE + hiHalf * 8;
    const float*    x0a  = xT + (d0a + laneN) * XSTRIDE;
    const float*    x0b  = xT + (d0b + laneN) * XSTRIDE;

    // xk fragments are i-invariant: load them once for the whole layer.
    v16h vb0[HKB], vb1[HKB];
#pragma unroll
    for (int q = 0; q < HKB; ++q) {
        const _Float16* pa = xka + q * 32;
        const _Float16* pb = xkb + q * 32;
        vb0[q] = cat16(*(const v8h*)pa, *(const v8h*)(pa + 16));
        vb1[q] = cat16(*(const v8h*)pb, *(const v8h*)(pb + 16));
    }

    v8f z0 = {}, z1 = {};
    for (int i = 0; i < M_FIELDS; ++i) {
        v8f y0 = {}, y1 = {};                  // fresh accumulators (inline-0 C)
#pragma unroll
        for (int q = 0; q < HKB; ++q) {
            const _Float16* wp = wrow + (i * HKB + q) * 32;
            const v16h va = cat16(*(const v8h*)wp, *(const v8h*)(wp + 16));
            y0 = wmma_f16(va, vb0[q], y0);
            y1 = wmma_f16(va, vb1[q], y1);
        }
        const float sa = x0a[i];               // ds_load (pairs combine)
        const float sb = x0b[i];
#pragma unroll
        for (int r = 0; r < 8; ++r) {          // f32 x0-scaling: v_fmac_f32
            z0[r] += sa * y0[r];
            z1[r] += sb * y1[r];
        }
    }

    finish_tile<STORE_H>(z0, hOut, osum, h0, d0a, lane);
    finish_tile<STORE_H>(z1, hOut, osum, h0, d0b, lane);
}

__global__ __launch_bounds__(THREADS)
void cin_fused_kernel(const float* __restrict__ X,
                      const _Float16* __restrict__ Wt,
                      float* __restrict__ out) {
    __shared__ float     xT[EMB * XSTRIDE];   // x[b] transposed f32 (x0 factors)
    __shared__ _Float16  xF[EMB * XSTRH];     // x[b] transposed f16 (xk rows)
    __shared__ _Float16  hA[EMB * HSTRIDE];   // h1 transposed, f16
    __shared__ _Float16  hB[EMB * HSTRIDE];   // h2 transposed, f16
    __shared__ float     osum[HN];

    const int b   = blockIdx.x;
    const int tid = threadIdx.x;

    // Zero pads (columns j>=39 of xF), then fill x.
    for (int idx = tid; idx < EMB * XSTRIDE; idx += THREADS) xT[idx] = 0.f;
    for (int idx = tid; idx < EMB * XSTRH;   idx += THREADS) xF[idx] = (_Float16)0.f;
    if (tid < HN) osum[tid] = 0.f;
    __syncthreads();
    for (int idx = tid; idx < M_FIELDS * EMB; idx += THREADS) {
        const int i = idx / EMB;
        const int d = idx % EMB;                 // consecutive d -> coalesced
        const float v = X[(b * M_FIELDS + i) * EMB + d];
        xT[d * XSTRIDE + i] = v;
        xF[d * XSTRH + i]   = (_Float16)v;
    }
    __syncthreads();

    // Layer 1: x (x) x -> h1.  Per-field K padded 39->64, zeros absorb pads.
    cin_layer<HKP0 / 32, KP0, XSTRH, true>(Wt + WT_OFF0, xT, xF, hA, osum, tid);
    __syncthreads();
    if (tid < HN) { out[b * 384 + tid] = osum[tid]; osum[tid] = 0.f; }
    __syncthreads();

    // Layer 2: x (x) h1 -> h2.  Per-field K = 128.
    cin_layer<HN / 32, K12, HSTRIDE, true>(Wt + WT_OFF1, xT, hA, hB, osum, tid);
    __syncthreads();
    if (tid < HN) { out[b * 384 + 128 + tid] = osum[tid]; osum[tid] = 0.f; }
    __syncthreads();

    // Layer 3: x (x) h2 -> d-sums only (h3 never materialized).
    cin_layer<HN / 32, K12, HSTRIDE, false>(Wt + WT_OFF2, xT, hB, nullptr, osum, tid);
    __syncthreads();
    if (tid < HN) out[b * 384 + 256 + tid] = osum[tid];
}

extern "C" void kernel_launch(void* const* d_in, const int* in_sizes, int n_in,
                              void* d_out, int out_size, void* d_ws, size_t ws_size,
                              hipStream_t stream) {
    const float* X  = (const float*)d_in[0];
    const float* W0 = (const float*)d_in[1];
    const float* W1 = (const float*)d_in[2];
    const float* W2 = (const float*)d_in[3];
    float* out = (float*)d_out;
    _Float16* Wt = (_Float16*)d_ws;              // needs WT_BYTES (~3.05 MB)

    // Pass 1: weight transpose + f16 convert (stream-ordered before pass 2).
    wt0_kernel<<<(KP0 * HN + 255) / 256, 256, 0, stream>>>(W0, Wt + WT_OFF0);
    wt12_kernel<<<(K12 * HN + 255) / 256, 256, 0, stream>>>(W1, Wt + WT_OFF1);
    wt12_kernel<<<(K12 * HN + 255) / 256, 256, 0, stream>>>(W2, Wt + WT_OFF2);

    // Pass 2: fused 3-layer CIN.
    const int B = in_sizes[0] / (M_FIELDS * EMB);   // 4096
    cin_fused_kernel<<<B, THREADS, 0, stream>>>(X, Wt, out);
}
